// RecursiveLSTM_19636590477959
// MI455X (gfx1250) — compile-verified, matrix operand order checked by reading, not run on hardware
//
#include <hip/hip_runtime.h>
#include <hip/hip_bf16.h>
#include <math.h>

// ---- model dims ----
#define D_VOCAB  32000
#define D_EMBED  256
#define D_HIDDEN 512
#define D_COND   512
#define T_MAX    128
#define BATCH    32
#define G4H      (4 * D_HIDDEN)   // 2048

// ---- WMMA fragment types (CDNA5 wave32) ----
typedef __attribute__((ext_vector_type(16))) __bf16       v16bf;
typedef __attribute__((ext_vector_type(8)))  float        v8f;
typedef __attribute__((ext_vector_type(4)))  unsigned int v4u;

union FragAB {
  v16bf bf;
  v4u   q[2];   // two 16-byte chunks: K {0..7} and K {16..23} (+8 for hi lanes)
};

__device__ __forceinline__ unsigned short f32_bf16(float f) {
  unsigned int u = __float_as_uint(f);
  u += 0x7FFFu + ((u >> 16) & 1u);   // round-to-nearest-even
  return (unsigned short)(u >> 16);
}

// order-preserving f32 -> u32 key; pack with ~idx so atomicMax ties pick lowest idx
__device__ __forceinline__ unsigned long long pack_argmax(float v, int idx) {
  unsigned int u   = __float_as_uint(v);
  unsigned int key = (u & 0x80000000u) ? ~u : (u | 0x80000000u);
  return ((unsigned long long)key << 32) | (unsigned int)(~idx);
}

// ---------- one-time: fp32 -> bf16 weight conversion ----------
__global__ void k_cvt_bf16(const float* __restrict__ src,
                           unsigned short* __restrict__ dst, int n) {
  int i = blockIdx.x * blockDim.x + threadIdx.x;
  if (i < n) dst[i] = f32_bf16(src[i]);
}

// ---------- one-time: h0 = cond@fc_h_w.T + b ; c0 = cond@fc_c_w.T + b ----------
__global__ void k_init(const float* __restrict__ cond,
                       const float* __restrict__ fhw, const float* __restrict__ fhb,
                       const float* __restrict__ fcw, const float* __restrict__ fcb,
                       unsigned short* __restrict__ h_bf, float* __restrict__ c,
                       int* __restrict__ tok) {
  int idx = blockIdx.x * blockDim.x + threadIdx.x;   // 32*512
  int b = idx >> 9, j = idx & 511;
  const float* cr = cond + (size_t)b * D_COND;
  const float* wh = fhw + (size_t)j * D_COND;
  const float* wc = fcw + (size_t)j * D_COND;
  float ah = fhb[j], ac = fcb[j];
  for (int k = 0; k < D_COND; ++k) { float cv = cr[k]; ah += cv * wh[k]; ac += cv * wc[k]; }
  h_bf[idx] = f32_bf16(ah);
  c[idx]    = ac;
  if (j == 0) tok[b] = 1;   // START_TOKEN
}

// ---------- per step: gates = [emb|h] @ [w_ih|w_hh].T + b_ih + b_hh ----------
// M=32 (2 tiles), N=2048 (128 tiles), K=768 (24 WMMA steps). 256 waves total.
__global__ void __launch_bounds__(256, 2)
k_gates(const unsigned short* __restrict__ emb_bf,   // [V,256] bf16
        const unsigned short* __restrict__ wih_bf,   // [2048,256] bf16
        const unsigned short* __restrict__ whh_bf,   // [2048,512] bf16
        const unsigned short* __restrict__ h_bf,     // [32,512] bf16
        const int* __restrict__ tok,
        const float* __restrict__ b_ih, const float* __restrict__ b_hh,
        float* __restrict__ gates)                   // [32,2048] f32
{
  const int lane  = threadIdx.x & 31;
  const int wave  = (blockIdx.x * blockDim.x + threadIdx.x) >> 5;  // 0..255
  const int ntile = wave & 127;
  const int mtile = wave >> 7;
  const int l15   = lane & 15;
  const int khalf = (lane >> 4) << 3;      // 0 (lanes 0-15) or 8 (lanes 16-31)
  const int m     = mtile * 16 + l15;
  const int n     = ntile * 16 + l15;

  const unsigned short* arow_e = emb_bf + (size_t)tok[m] * D_EMBED;
  const unsigned short* arow_h = h_bf   + (size_t)m      * D_HIDDEN;
  const unsigned short* brow_i = wih_bf + (size_t)n      * D_EMBED;
  const unsigned short* brow_h = whh_bf + (size_t)n      * D_HIDDEN;

  v8f acc = {};
  // K segment 1: embedding part (K = 0..255)
  #pragma unroll
  for (int kb = 0; kb < D_EMBED; kb += 32) {
    FragAB a, b;
    a.q[0] = *reinterpret_cast<const v4u*>(arow_e + kb + khalf);
    a.q[1] = *reinterpret_cast<const v4u*>(arow_e + kb + 16 + khalf);
    b.q[0] = *reinterpret_cast<const v4u*>(brow_i + kb + khalf);
    b.q[1] = *reinterpret_cast<const v4u*>(brow_i + kb + 16 + khalf);
    acc = __builtin_amdgcn_wmma_f32_16x16x32_bf16(false, a.bf, false, b.bf,
                                                  (short)0, acc, false, false);
  }
  // K segment 2: hidden part (K = 256..767)
  #pragma unroll
  for (int kb = 0; kb < D_HIDDEN; kb += 32) {
    FragAB a, b;
    a.q[0] = *reinterpret_cast<const v4u*>(arow_h + kb + khalf);
    a.q[1] = *reinterpret_cast<const v4u*>(arow_h + kb + 16 + khalf);
    b.q[0] = *reinterpret_cast<const v4u*>(brow_h + kb + khalf);
    b.q[1] = *reinterpret_cast<const v4u*>(brow_h + kb + 16 + khalf);
    acc = __builtin_amdgcn_wmma_f32_16x16x32_bf16(false, a.bf, false, b.bf,
                                                  (short)0, acc, false, false);
  }

  const float bias  = b_ih[n] + b_hh[n];
  const int   mbase = mtile * 16 + ((lane >> 4) << 3);  // D: lanes16-31 hold M+8
  #pragma unroll
  for (int r = 0; r < 8; ++r)
    gates[(size_t)(mbase + r) * G4H + n] = acc[r] + bias;
}

// ---------- per step: LSTM cell pointwise (+ reset argmax slots) ----------
__global__ void k_cell(const float* __restrict__ gates, float* __restrict__ c,
                       unsigned short* __restrict__ h_bf,
                       unsigned long long* __restrict__ tok_pack) {
  int idx = blockIdx.x * blockDim.x + threadIdx.x;  // 32*512
  int b = idx >> 9, j = idx & 511;
  const float* g = gates + (size_t)b * G4H;
  float xi = g[j], xf = g[j + 512], xg = g[j + 1024], xo = g[j + 1536];
  float i = 1.f / (1.f + __expf(-xi));
  float f = 1.f / (1.f + __expf(-xf));
  float gg = tanhf(xg);
  float o = 1.f / (1.f + __expf(-xo));
  float cn = f * c[idx] + i * gg;
  c[idx] = cn;
  h_bf[idx] = f32_bf16(o * tanhf(cn));
  if (idx < BATCH) tok_pack[idx] = 0ull;   // floor for upcoming atomicMax argmax
}

// ---------- per step: logits = h @ logits_w.T + b, fused argmax ----------
// M=32 (2 tiles), N=32000 (2000 tiles), K=512 (16 WMMA steps). 4000 waves.
__global__ void __launch_bounds__(256, 2)
k_logits(const unsigned short* __restrict__ lw_bf,   // [32000,512] bf16 (L2-resident)
         const unsigned short* __restrict__ h_bf,    // [32,512] bf16
         const float* __restrict__ lb,                // [32000]
         unsigned long long* __restrict__ tok_pack,   // [32] packed (key<<32 | ~idx)
         float* __restrict__ out_logits,              // d_out logits [32][32000][128]
         int t)
{
  const int lane  = threadIdx.x & 31;
  const int wave  = (blockIdx.x * blockDim.x + threadIdx.x) >> 5;  // 0..3999
  const int mtile = wave & 1;
  const int ntile = wave >> 1;
  const int l15   = lane & 15;
  const int khalf = (lane >> 4) << 3;
  const int m     = mtile * 16 + l15;
  const int n     = ntile * 16 + l15;

  const unsigned short* arow = h_bf  + (size_t)m * D_HIDDEN;
  const unsigned short* brow = lw_bf + (size_t)n * D_HIDDEN;

  v8f acc = {};
  #pragma unroll 4
  for (int kb = 0; kb < D_HIDDEN; kb += 32) {
    FragAB a, b;
    a.q[0] = *reinterpret_cast<const v4u*>(arow + kb + khalf);
    a.q[1] = *reinterpret_cast<const v4u*>(arow + kb + 16 + khalf);
    b.q[0] = *reinterpret_cast<const v4u*>(brow + kb + khalf);
    b.q[1] = *reinterpret_cast<const v4u*>(brow + kb + 16 + khalf);
    __builtin_prefetch(brow + kb + 64, 0, 0);   // global_prefetch_b8, next K chunk
    acc = __builtin_amdgcn_wmma_f32_16x16x32_bf16(false, a.bf, false, b.bf,
                                                  (short)0, acc, false, false);
  }

  const float bias  = lb[n];
  const int   mbase = mtile * 16 + ((lane >> 4) << 3);
  #pragma unroll
  for (int r = 0; r < 8; ++r) {
    float v  = acc[r] + bias;
    int  mm  = mbase + r;
    out_logits[((size_t)mm * D_VOCAB + n) * T_MAX + t] = v;   // [B,V,T] layout

    // in-register row argmax: reduce across the 16-lane half holding row mm
    unsigned long long p = pack_argmax(v, n);
    #pragma unroll
    for (int s = 1; s < 16; s <<= 1) {
      unsigned long long q = __shfl_xor(p, s, 32);
      p = (q > p) ? q : p;
    }
    if (l15 == 0) atomicMax(&tok_pack[mm], p);   // global_atomic_max_u64
  }
}

// ---------- per step: decode winning token + write caption ----------
__global__ void k_pick(const unsigned long long* __restrict__ tok_pack,
                       int* __restrict__ tok, float* __restrict__ out_caps, int t) {
  int b = threadIdx.x;                 // 32 threads
  if (b < BATCH) {
    int idx = (int)(~(unsigned int)(tok_pack[b] & 0xffffffffull));
    tok[b] = idx;
    out_caps[(size_t)b * T_MAX + t] = (float)idx;
  }
}

// =====================================================================
extern "C" void kernel_launch(void* const* d_in, const int* in_sizes, int n_in,
                              void* d_out, int out_size, void* d_ws, size_t ws_size,
                              hipStream_t stream) {
  const float* condition = (const float*)d_in[0];
  const float* embed_w   = (const float*)d_in[1];
  const float* fc_h_w    = (const float*)d_in[2];
  const float* fc_h_b    = (const float*)d_in[3];
  const float* fc_c_w    = (const float*)d_in[4];
  const float* fc_c_b    = (const float*)d_in[5];
  const float* w_ih      = (const float*)d_in[6];
  const float* w_hh      = (const float*)d_in[7];
  const float* b_ih      = (const float*)d_in[8];
  const float* b_hh      = (const float*)d_in[9];
  const float* logits_w  = (const float*)d_in[10];
  const float* logits_b  = (const float*)d_in[11];

  // ---- workspace carve-out (~53 MB, 256B aligned) ----
  char* ws = (char*)d_ws;
  size_t off = 0;
  auto carve = [&](size_t bytes) -> char* {
    char* p = ws + off;
    off = (off + bytes + 255) & ~(size_t)255;
    return p;
  };
  unsigned short* lw_bf  = (unsigned short*)carve((size_t)D_VOCAB * D_HIDDEN * 2);
  unsigned short* emb_bf = (unsigned short*)carve((size_t)D_VOCAB * D_EMBED  * 2);
  unsigned short* wih_bf = (unsigned short*)carve((size_t)G4H     * D_EMBED  * 2);
  unsigned short* whh_bf = (unsigned short*)carve((size_t)G4H     * D_HIDDEN * 2);
  unsigned short* h_bf   = (unsigned short*)carve((size_t)BATCH   * D_HIDDEN * 2);
  float* c_st            = (float*)carve((size_t)BATCH * D_HIDDEN * 4);
  float* gates           = (float*)carve((size_t)BATCH * G4H     * 4);
  unsigned long long* tok_pack = (unsigned long long*)carve(BATCH * 8);
  int*   tok             = (int*)carve(BATCH * 4);

  float* out_caps   = (float*)d_out;                       // [B,T] token ids
  float* out_logits = out_caps + (size_t)BATCH * T_MAX;    // [B,V,T]

  // ---- one-time bf16 weight conversion (then L2-resident for 128 steps) ----
  { int n = D_VOCAB * D_HIDDEN; k_cvt_bf16<<<(n + 255) / 256, 256, 0, stream>>>(logits_w, lw_bf, n); }
  { int n = D_VOCAB * D_EMBED;  k_cvt_bf16<<<(n + 255) / 256, 256, 0, stream>>>(embed_w, emb_bf, n); }
  { int n = G4H * D_EMBED;      k_cvt_bf16<<<(n + 255) / 256, 256, 0, stream>>>(w_ih, wih_bf, n); }
  { int n = G4H * D_HIDDEN;     k_cvt_bf16<<<(n + 255) / 256, 256, 0, stream>>>(w_hh, whh_bf, n); }

  k_init<<<64, 256, 0, stream>>>(condition, fc_h_w, fc_h_b, fc_c_w, fc_c_b,
                                 h_bf, c_st, tok);

  // ---- sequential decode: 4 kernels per step (pick is trivial), graph-captured ----
  for (int t = 0; t < T_MAX; ++t) {
    k_gates <<<32,  256, 0, stream>>>(emb_bf, wih_bf, whh_bf, h_bf, tok,
                                      b_ih, b_hh, gates);
    k_cell  <<<64,  256, 0, stream>>>(gates, c_st, h_bf, tok_pack);
    k_logits<<<500, 256, 0, stream>>>(lw_bf, h_bf, logits_b,
                                      tok_pack, out_logits, t);
    k_pick  <<<1,   32,  0, stream>>>(tok_pack, tok, out_caps, t);
  }
}
